// SpatialAwareMemory_3032246911417
// MI455X (gfx1250) — matrix operand order repaired and checked by reading, not compiled, not run on hardware
//
#include <hip/hip_runtime.h>

#define NPTS    65536
#define DIM     256
#define MEMROWS 2048
#define SCALE   0.0625f   // 1/sqrt(256)

typedef __attribute__((ext_vector_type(16))) __bf16 v16bf;
typedef __attribute__((ext_vector_type(8)))  float  v8f;
typedef __attribute__((ext_vector_type(4)))  unsigned int u32x4;
typedef __attribute__((ext_vector_type(8)))  int i32x8;
typedef __attribute__((ext_vector_type(4)))  int i32x4;

union BF16x16 {
  v16bf v;
  unsigned short u[16];
  uint4 q[2];
};

__device__ __forceinline__ unsigned short f2bf(float f) {
  union { float f; unsigned int u; } x; x.f = f;
  unsigned int r = x.u + 0x7FFFu + ((x.u >> 16) & 1u);   // round-to-nearest-even
  return (unsigned short)(r >> 16);
}

// ---------------- prep: coords mean -> 2-layer MLP -> emb[256] ----------------
__global__ void prep_kernel(const float* __restrict__ coords,
                            const float* __restrict__ w1,
                            const float* __restrict__ b1,
                            const float* __restrict__ w2,
                            const float* __restrict__ b2,
                            float* __restrict__ emb) {
  __shared__ float sx[256], sy[256], h[256];
  int t = threadIdx.x;
  float ax = 0.f, ay = 0.f;
  for (int i = t; i < NPTS; i += 256) { ax += coords[2*i]; ay += coords[2*i+1]; }
  sx[t] = ax; sy[t] = ay; __syncthreads();
  for (int s = 128; s > 0; s >>= 1) {
    if (t < s) { sx[t] += sx[t+s]; sy[t] += sy[t+s]; }
    __syncthreads();
  }
  float mx = sx[0] * (1.0f/NPTS), my = sy[0] * (1.0f/NPTS);
  float hv = mx * w1[2*t] + my * w1[2*t+1] + b1[t];
  h[t] = hv > 0.f ? hv : 0.f;
  __syncthreads();
  float a = b2[t];
  for (int j = 0; j < DIM; ++j) a += h[j] * w2[t*DIM + j];
  emb[t] = a;
}

// ------- prep2: memory f32 -> bf16 row-major + bf16 transposed (in ws) -------
__global__ __launch_bounds__(256) void convert_kernel(
    const float* __restrict__ msrc,
    unsigned short* __restrict__ mem_bf,
    unsigned short* __restrict__ memT_bf) {
  __shared__ unsigned short tile[32][33];
  const int tx = threadIdx.x & 31, ty = threadIdx.x >> 5;   // 32 x 8
  const int r0 = blockIdx.x * 32, c0 = blockIdx.y * 32;
#pragma unroll
  for (int i = 0; i < 32; i += 8) {
    int r = r0 + ty + i, c = c0 + tx;
    unsigned short b = f2bf(msrc[r * DIM + c]);
    mem_bf[r * DIM + c] = b;
    tile[ty + i][tx] = b;
  }
  __syncthreads();
#pragma unroll
  for (int i = 0; i < 32; i += 8) {
    int c = c0 + ty + i, r = r0 + tx;
    memT_bf[c * MEMROWS + r] = tile[tx][ty + i];
  }
}

// ---------------- fused flash-attention over fixed memory K/V ----------------
// LDS strides (halfwords): byte stride % 16 == 0 (aligned ds_load_b128) and
// 16 consecutive rows land on distinct banks.
#define BROWS 280   // 32 x 280 halves : row-major chunk (QK B-tiles); 512B data + 48B pad
#define VTS   40    // 256 x 40 halves : transposed chunk (PV B-tiles); 64B data + 16B pad
#define PBS   40    // per-wave P bounce buffer row stride

// Issue one 2D TDM tile load (global bf16 -> LDS), with LDS row padding.
__device__ __forceinline__ void tdm_load_2d(unsigned lds_off,
                                            unsigned long long gaddr,
                                            unsigned tile_d0, unsigned tile_d1,
                                            unsigned tens_d0, unsigned tens_d1,
                                            unsigned long long stride0,
                                            unsigned pad_interval_code,
                                            unsigned pad_amount_code) {
  u32x4 g0;
  g0.x = 1u;                                               // count=1, user mode
  g0.y = lds_off;                                          // LDS byte address
  g0.z = (unsigned)(gaddr & 0xFFFFFFFFull);                // global_addr[31:0]
  g0.w = (unsigned)((gaddr >> 32) & 0x1FFFFFFull) | (2u << 30);  // [56:32] | type=2
  i32x8 g1;
  g1[0] = (int)((1u << 16)                                 // data_size = 2 bytes
              | (1u << 20)                                 // pad_enable
              | (pad_interval_code << 22)
              | (pad_amount_code << 25));                  // workgroup_mask = 0
  g1[1] = (int)((tens_d0 & 0xFFFFu) << 16);                // tensor_dim0[15:0]
  g1[2] = (int)((tens_d0 >> 16) | ((tens_d1 & 0xFFFFu) << 16));
  g1[3] = (int)((tens_d1 >> 16) | ((tile_d0 & 0xFFFFu) << 16));
  g1[4] = (int)(tile_d1 & 0xFFFFu);                        // tile_dim2 = 0
  g1[5] = (int)(unsigned)(stride0 & 0xFFFFFFFFull);        // tensor_dim0_stride
  g1[6] = (int)(unsigned)((stride0 >> 32) & 0xFFFFu);      // stride1 = 0
  g1[7] = 0;
  i32x4 gz = {0, 0, 0, 0};                                 // groups 2/3: 2D tensor
#if defined(__clang_major__) && __clang_major__ >= 23
  i32x8 gz8 = {0, 0, 0, 0, 0, 0, 0, 0};
  __builtin_amdgcn_tensor_load_to_lds(g0, g1, gz, gz, gz8, 0);
#else
  __builtin_amdgcn_tensor_load_to_lds(g0, g1, gz, gz, 0);
#endif
}

__global__ __launch_bounds__(128) void attn_kernel(
    const float* __restrict__ features,
    const unsigned short* __restrict__ mem_bf,
    const unsigned short* __restrict__ memT_bf,
    const float* __restrict__ emb,
    float* __restrict__ out) {
  __shared__ __attribute__((aligned(16))) unsigned short Brow[2][32 * BROWS];
  __shared__ __attribute__((aligned(16))) unsigned short VT[2][DIM * VTS];
  __shared__ __attribute__((aligned(16))) unsigned short Pb[4][16 * PBS];
  __shared__ __attribute__((aligned(16))) float semb[DIM];

  const int tid  = threadIdx.x;          // 0..127, 4 waves
  const int wave = tid >> 5;
  const int lane = tid & 31;
  const int half = lane >> 4;            // 0: lanes 0-15, 1: lanes 16-31
  const int lr   = lane & 15;
  const int rowbase = blockIdx.x * 64 + wave * 16;
  const int row  = rowbase + lr;

  semb[tid]       = emb[tid];
  semb[tid + 128] = emb[tid + 128];

  // Kick off chunk 0 DMA while we build z.
  if (wave == 0) {
    tdm_load_2d((unsigned)(unsigned long long)(uintptr_t)&Brow[0][0],
                (unsigned long long)(uintptr_t)mem_bf,
                /*tile*/ 256, 32, /*tensor*/ 256, MEMROWS, /*stride0*/ 256,
                /*pad: every 128 dwords add 12 dwords*/ 6, 11);
    tdm_load_2d((unsigned)(unsigned long long)(uintptr_t)&VT[0][0],
                (unsigned long long)(uintptr_t)memT_bf,
                /*tile*/ 32, 256, /*tensor*/ MEMROWS, 256, /*stride0*/ MEMROWS,
                /*pad: every 16 dwords add 4 dwords*/ 3, 3);
  }
  __syncthreads();   // semb visible

  // z = features(row) + emb, packed in WMMA A-layout (16x32 bf16 per tile).
  // half=0 lanes hold K {0..7,16..23} (+32*t), half=1 hold K {8..15,24..31}.
  BF16x16 za[8];
  const float* fr = features + (size_t)row * DIM;
#pragma unroll
  for (int t = 0; t < 8; ++t) {
    int k0 = t*32 + half*8;
    float4 fa = ((const float4*)(fr + k0))[0];
    float4 fb = ((const float4*)(fr + k0))[1];
    float4 ga = ((const float4*)(fr + k0 + 16))[0];
    float4 gb = ((const float4*)(fr + k0 + 16))[1];
    float4 ea = ((const float4*)(semb + k0))[0];
    float4 eb = ((const float4*)(semb + k0))[1];
    float4 ha = ((const float4*)(semb + k0 + 16))[0];
    float4 hb = ((const float4*)(semb + k0 + 16))[1];
    za[t].u[0]  = f2bf(fa.x + ea.x); za[t].u[1]  = f2bf(fa.y + ea.y);
    za[t].u[2]  = f2bf(fa.z + ea.z); za[t].u[3]  = f2bf(fa.w + ea.w);
    za[t].u[4]  = f2bf(fb.x + eb.x); za[t].u[5]  = f2bf(fb.y + eb.y);
    za[t].u[6]  = f2bf(fb.z + eb.z); za[t].u[7]  = f2bf(fb.w + eb.w);
    za[t].u[8]  = f2bf(ga.x + ha.x); za[t].u[9]  = f2bf(ga.y + ha.y);
    za[t].u[10] = f2bf(ga.z + ha.z); za[t].u[11] = f2bf(ga.w + ha.w);
    za[t].u[12] = f2bf(gb.x + hb.x); za[t].u[13] = f2bf(gb.y + hb.y);
    za[t].u[14] = f2bf(gb.z + hb.z); za[t].u[15] = f2bf(gb.w + hb.w);
  }

  const v8f zero8 = {0.f,0.f,0.f,0.f,0.f,0.f,0.f,0.f};
  v8f acc[16];                      // O[16 rows][256 dims] as 16 C-tiles
#pragma unroll
  for (int t = 0; t < 16; ++t) acc[t] = zero8;
  float mrow[8], lrow[8];           // per-row running max / sum (rows r+8*half)
#pragma unroll
  for (int r = 0; r < 8; ++r) { mrow[r] = -1e30f; lrow[r] = 0.f; }

  for (int chunk = 0; chunk < MEMROWS/32; ++chunk) {
    const int buf = chunk & 1;
    __syncthreads();  // all waves done reading buf^1 (chunk-1) -> safe to refill
    if (wave == 0) {
      if (chunk + 1 < MEMROWS/32) {
        const int nbuf = buf ^ 1;
        tdm_load_2d((unsigned)(unsigned long long)(uintptr_t)&Brow[nbuf][0],
                    (unsigned long long)(uintptr_t)(mem_bf + (size_t)(chunk+1)*32*DIM),
                    256, 32, 256, MEMROWS, 256, 6, 11);
        tdm_load_2d((unsigned)(unsigned long long)(uintptr_t)&VT[nbuf][0],
                    (unsigned long long)(uintptr_t)(memT_bf + (size_t)(chunk+1)*32),
                    32, 256, MEMROWS, 256, MEMROWS, 3, 3);
        // TENSORcnt decrements in order: <=2 outstanding => chunk's pair is done.
        __builtin_amdgcn_s_wait_tensorcnt(2);
      } else {
        __builtin_amdgcn_s_wait_tensorcnt(0);
      }
    }
    __syncthreads();  // buf is ready for every wave

    // ---- QK: S[16x32] = z[16x256] @ memT tiles, f32 accum ----
    v8f S0 = zero8, S1 = zero8;
#pragma unroll
    for (int kt = 0; kt < 8; ++kt) {
      int kh = kt*32 + half*8;
      BF16x16 b0, b1;
      const uint4* p0 = (const uint4*)&Brow[buf][lr * BROWS + kh];
      const uint4* p1 = (const uint4*)&Brow[buf][(16 + lr) * BROWS + kh];
      b0.q[0] = p0[0]; b0.q[1] = p0[2];
      b1.q[0] = p1[0]; b1.q[1] = p1[2];
      S0 = __builtin_amdgcn_wmma_f32_16x16x32_bf16(false, za[kt].v, false, b0.v,
                                                   (short)0, S0, false, false);
      S1 = __builtin_amdgcn_wmma_f32_16x16x32_bf16(false, za[kt].v, false, b1.v,
                                                   (short)0, S1, false, false);
    }
#pragma unroll
    for (int r = 0; r < 8; ++r) { S0[r] *= SCALE; S1[r] *= SCALE; }

    // ---- online softmax stats (row reductions across 16-lane groups) ----
    float rmax[8];
#pragma unroll
    for (int r = 0; r < 8; ++r) rmax[r] = fmaxf(S0[r], S1[r]);
#pragma unroll
    for (int m = 1; m < 16; m <<= 1)
#pragma unroll
      for (int r = 0; r < 8; ++r)
        rmax[r] = fmaxf(rmax[r], __shfl_xor(rmax[r], m, 32));
    float corr[8];
#pragma unroll
    for (int r = 0; r < 8; ++r) {
      float mn = fmaxf(mrow[r], rmax[r]);
      corr[r] = __expf(mrow[r] - mn);
      mrow[r] = mn;
    }
    float rsum[8];
#pragma unroll
    for (int r = 0; r < 8; ++r) {
      S0[r] = __expf(S0[r] - mrow[r]);
      S1[r] = __expf(S1[r] - mrow[r]);
      rsum[r] = S0[r] + S1[r];
    }
#pragma unroll
    for (int m = 1; m < 16; m <<= 1)
#pragma unroll
      for (int r = 0; r < 8; ++r)
        rsum[r] += __shfl_xor(rsum[r], m, 32);
#pragma unroll
    for (int r = 0; r < 8; ++r) lrow[r] = lrow[r] * corr[r] + rsum[r];
#pragma unroll
    for (int t = 0; t < 16; ++t)
#pragma unroll
      for (int r = 0; r < 8; ++r)
        acc[t][r] *= corr[r];

    // ---- P: C-layout -> A-layout relayout, wave-private LDS bounce.
    // DS ops are in-order per wave, so only a wave-level reordering fence
    // is needed (no workgroup barrier).
    unsigned short* pb = Pb[wave];
#pragma unroll
    for (int r = 0; r < 8; ++r) {
      int prow = r + half*8;
      pb[prow * PBS + lr]      = f2bf(S0[r]);
      pb[prow * PBS + 16 + lr] = f2bf(S1[r]);
    }
    __builtin_amdgcn_fence(__ATOMIC_RELEASE, "wavefront");
    __builtin_amdgcn_wave_barrier();
    __builtin_amdgcn_fence(__ATOMIC_ACQUIRE, "wavefront");
    BF16x16 ap;
    {
      const uint4* q0 = (const uint4*)&pb[lr * PBS + half*8];
      const uint4* q1 = (const uint4*)&pb[lr * PBS + half*8 + 16];
      ap.q[0] = q0[0]; ap.q[1] = q1[0];
    }

    // ---- PV: O += P[16x32] @ V[32x256] ----
#pragma unroll
    for (int nt = 0; nt < 16; ++nt) {
      BF16x16 bv;
      const uint4* q = (const uint4*)&VT[buf][(nt*16 + lr) * VTS + half*8];
      bv.q[0] = q[0]; bv.q[1] = q[2];
      acc[nt] = __builtin_amdgcn_wmma_f32_16x16x32_bf16(false, ap.v, false, bv.v,
                                                        (short)0, acc[nt], false, false);
    }
  }

  // ---- epilogue: normalize by row sum and store ----
  float rl[8];
#pragma unroll
  for (int r = 0; r < 8; ++r) rl[r] = 1.0f / lrow[r];
#pragma unroll
  for (int nt = 0; nt < 16; ++nt)
#pragma unroll
    for (int r = 0; r < 8; ++r) {
      int orow = rowbase + r + half*8;
      out[(size_t)orow * DIM + nt*16 + lr] = acc[nt][r] * rl[r];
    }
}

extern "C" void kernel_launch(void* const* d_in, const int* in_sizes, int n_in,
                              void* d_out, int out_size, void* d_ws, size_t ws_size,
                              hipStream_t stream) {
  const float* features = (const float*)d_in[0];
  const float* coords   = (const float*)d_in[1];
  const float* memory   = (const float*)d_in[2];
  const float* w1       = (const float*)d_in[3];
  const float* b1       = (const float*)d_in[4];
  const float* w2       = (const float*)d_in[5];
  const float* b2       = (const float*)d_in[6];
  float* out = (float*)d_out;

  // workspace layout: emb f32[256] | mem_bf bf16[2048*256] | memT_bf bf16[256*2048]
  float* emb = (float*)d_ws;
  unsigned short* mem_bf  = (unsigned short*)((char*)d_ws + 1024);
  unsigned short* memT_bf = (unsigned short*)((char*)d_ws + 1024 + MEMROWS * DIM * 2);

  prep_kernel<<<1, 256, 0, stream>>>(coords, w1, b1, w2, b2, emb);
  convert_kernel<<<dim3(MEMROWS/32, DIM/32), 256, 0, stream>>>(memory, mem_bf, memT_bf);
  attn_kernel<<<NPTS / 64, 128, 0, stream>>>(features, mem_bf, memT_bf, emb, out);
}
